// FoldingNet_Encoder_18038862643366
// MI455X (gfx1250) — matrix operand order repaired and checked by reading, not compiled, not run on hardware
//
#include <hip/hip_runtime.h>
#include <hip/hip_bf16.h>
#include <stdint.h>

// FoldingNet encoder for MI455X (gfx1250, wave32).
// knn(async-LDS staging) -> cov/concat(bf16,padded) -> MLP1(wmma) -> maxpool
//  -> graph1(wmma) -> maxpool -> graph2(wmma, async double-buffered weights,
//     fused global-max) -> tail MLP2.

#define DEV __device__ __forceinline__

typedef __attribute__((ext_vector_type(16))) __bf16 bf16x16;
typedef __attribute__((ext_vector_type(8)))  float  f32x8;

static constexpr int Bsz  = 8;
static constexpr int Npts = 4096;
static constexpr int Mtot = Bsz * Npts;   // 32768

DEV f32x8 wmma_bf16(bf16x16 a, bf16x16 b, f32x8 c) {
  return __builtin_amdgcn_wmma_f32_16x16x32_bf16(
      false, a, false, b, (short)0, c, false, false);
}

// async global->LDS copy, 16B per lane (ASYNCcnt-tracked)
DEV void async_copy_b128(unsigned lds_off, const void* gaddr) {
  asm volatile("global_load_async_to_lds_b128 %0, %1, off"
               :: "v"(lds_off), "v"(gaddr) : "memory");
}
DEV void wait_async0() {
  asm volatile("s_wait_asynccnt 0x0" ::: "memory");
}
DEV unsigned lds_addr32(const void* p) {   // flat shared addr -> LDS byte offset
  return (unsigned)(size_t)p;
}

// ---- fragment element maps (ISA 7.12.2, 16-bit A 16x32 / B 32x16) ----
DEV int b_kmap(int e, int half) {
  int v = e >> 1, r = e & 1;
  return (v < 4) ? (2 * v + r + 8 * half) : (16 + 2 * (v - 4) + r + 8 * half);
}

// A fragment from bf16 row-major [M,ld]: 8 dword loads (K-pairs contiguous)
DEV bf16x16 load_a_bf16(const __bf16* src, int row0, int ld, int kb, int lane) {
  const unsigned* p = (const unsigned*)(src + (size_t)(row0 + (lane & 15)) * ld + kb);
  int half = lane >> 4;
  union { bf16x16 f; unsigned d[8]; } u;
#pragma unroll
  for (int w = 0; w < 8; ++w) {
    int k = (w < 4) ? (2 * w + 8 * half) : (16 + 2 * (w - 4) + 8 * half);
    u.d[w] = p[k >> 1];
  }
  return u.f;
}

// A fragment from wave-private LDS tile [16][ldst] bf16 (dword loads)
DEV bf16x16 load_a_lds(const __bf16* t, int ldst, int kb, int lane) {
  const unsigned* p = (const unsigned*)(t + (lane & 15) * ldst + kb);
  int half = lane >> 4;
  union { bf16x16 f; unsigned d[8]; } u;
#pragma unroll
  for (int w = 0; w < 8; ++w) {
    int k = (w < 4) ? (2 * w + 8 * half) : (16 + 2 * (w - 4) + 8 * half);
    u.d[w] = p[k >> 1];
  }
  return u.f;
}

// Packed-weight B fragment from global: one 32B vector load per lane
DEV bf16x16 load_b_packed(const __bf16* base, int g, int nKC, int kc, int lane) {
  return *(const bf16x16*)(base + (((size_t)(g * nKC + kc) * 32) + lane) * 16);
}
// Packed-weight B fragment from LDS staging buffer
DEV bf16x16 load_b_lds(const __bf16* buf, int kc, int lane) {
  return *(const bf16x16*)(buf + ((size_t)(kc * 32 + lane)) * 16);
}

// D (row m=r+8*half, col n=lane%16) -> LDS bf16 tile, + bias (+relu)
DEV void stage_acc(__bf16* t, int ldst, f32x8 acc, int n0, int lane,
                   const float* bias, bool relu) {
  int n = n0 + (lane & 15), half = lane >> 4;
  float bb = bias[n];
#pragma unroll
  for (int r = 0; r < 8; ++r) {
    float x = acc[r] + bb;
    if (relu) x = fmaxf(x, 0.0f);
    t[(r + 8 * half) * ldst + n] = (__bf16)x;
  }
}

// Cooperative row-major tile store: NC cols, b128 chunks per lane
template <int NC>
DEV void store_tile_rows(const __bf16* t, int ldst, __bf16* dst, int ld, int lane) {
  int row = lane & 15, ch = lane >> 4;
  const uint4* s = (const uint4*)(t + row * ldst + ch * (NC / 2));
  uint4* d = (uint4*)(dst + (size_t)row * ld + ch * (NC / 2));
#pragma unroll
  for (int q = 0; q < NC / 16; ++q) d[q] = s[q];
}

// ---- bf16 scalar helpers ----
DEV float bflo(unsigned u) { return __uint_as_float(u << 16); }
DEV float bfhi(unsigned u) { return __uint_as_float(u & 0xFFFF0000u); }
DEV unsigned f2bf(float f) {
  unsigned u = __float_as_uint(f);
  u += 0x7FFFu + ((u >> 16) & 1u);
  return u >> 16;
}

DEV void atomicMaxFloatShared(int* addr, float v) {
  int iv = __float_as_int(v);
  if (iv >= 0) atomicMax(addr, iv);
  else atomicMin((unsigned int*)addr, (unsigned int)iv);
}

// ====== Kernel 0: pack weight matrix into fragment-native bf16 layout =======
__global__ void pack_w_kernel(const float* __restrict__ W, __bf16* __restrict__ dst,
                              int Cin, int nG, int nKC) {
  int t = blockIdx.x * 256 + threadIdx.x;
  if (t >= nG * nKC * 32) return;
  int lane = t & 31, fragIdx = t >> 5;
  int kc = fragIdx % nKC, g = fragIdx / nKC;
  int n = g * 16 + (lane & 15), half = lane >> 4;
  __bf16* o = dst + (size_t)t * 16;
#pragma unroll
  for (int e = 0; e < 16; ++e) {
    int k = kc * 32 + b_kmap(e, half);
    float x = (k < Cin) ? W[(size_t)n * Cin + k] : 0.0f;
    o[e] = (__bf16)x;
  }
}

// ================= Kernel 1: KNN (top-16 by squared distance) ===============
__global__ void knn_kernel(const float* __restrict__ pts, int* __restrict__ idxout) {
  __shared__ alignas(16) float P3[Npts * 3];   // 48KB AoS, async-staged
  int b = blockIdx.y, tid = threadIdx.x;
  const float* Pg = pts + (size_t)b * Npts * 3;
  unsigned lbase = lds_addr32(P3);
#pragma unroll
  for (int it = 0; it < 12; ++it) {            // 3072 x 16B chunks
    int chunk = it * 256 + tid;
    async_copy_b128(lbase + chunk * 16, (const char*)Pg + (size_t)chunk * 16);
  }
  wait_async0();
  __syncthreads();
  int i = blockIdx.x * 256 + tid;
  float qx = P3[i * 3 + 0], qy = P3[i * 3 + 1], qz = P3[i * 3 + 2];
  float d[16]; int id[16];
#pragma unroll
  for (int s = 0; s < 16; ++s) { d[s] = 3.4e38f; id[s] = 0; }
  for (int j = 0; j < Npts; ++j) {             // uniform j -> LDS broadcast
    float dx = P3[j * 3 + 0] - qx, dy = P3[j * 3 + 1] - qy, dz = P3[j * 3 + 2] - qz;
    float dd = fmaf(dx, dx, fmaf(dy, dy, dz * dz));
    if (dd < d[15]) {
      d[15] = dd; id[15] = j;
#pragma unroll
      for (int s = 14; s >= 0; --s) {
        if (d[s] > d[s + 1]) {
          float td = d[s]; d[s] = d[s + 1]; d[s + 1] = td;
          int ti = id[s]; id[s] = id[s + 1]; id[s + 1] = ti;
        }
      }
    }
  }
  int* o = idxout + ((size_t)b * Npts + i) * 16;
#pragma unroll
  for (int s = 0; s < 16; ++s) o[s] = id[s];
}

// ====== Kernel 2: covariance + concat -> x0p bf16 [M,32] (zero-padded) ======
__global__ void cov_kernel(const float* __restrict__ pts, const int* __restrict__ idx,
                           __bf16* __restrict__ x0p) {
  int gi = blockIdx.x * 256 + threadIdx.x;
  if (gi >= Mtot) return;
  int b = gi >> 12, li = gi & (Npts - 1);
  const float* P = pts + (size_t)b * Npts * 3;
  const int* nb = idx + (size_t)gi * 16;
  float nx[16], ny[16], nz[16], mx = 0, my = 0, mz = 0;
#pragma unroll
  for (int k = 0; k < 16; ++k) {
    int j = nb[k];
    nx[k] = P[j * 3]; ny[k] = P[j * 3 + 1]; nz[k] = P[j * 3 + 2];
    mx += nx[k]; my += ny[k]; mz += nz[k];
  }
  mx *= 0.0625f; my *= 0.0625f; mz *= 0.0625f;
  float c00 = 0, c01 = 0, c02 = 0, c11 = 0, c12 = 0, c22 = 0;
#pragma unroll
  for (int k = 0; k < 16; ++k) {
    float ax = nx[k] - mx, ay = ny[k] - my, az = nz[k] - mz;
    c00 = fmaf(ax, ax, c00); c01 = fmaf(ax, ay, c01); c02 = fmaf(ax, az, c02);
    c11 = fmaf(ay, ay, c11); c12 = fmaf(ay, az, c12); c22 = fmaf(az, az, c22);
  }
  float v12[12] = { P[li * 3], P[li * 3 + 1], P[li * 3 + 2],
                    c00, c01, c02, c01, c11, c12, c02, c12, c22 };
  unsigned ow[16];
#pragma unroll
  for (int q = 0; q < 6; ++q) ow[q] = f2bf(v12[2 * q]) | (f2bf(v12[2 * q + 1]) << 16);
#pragma unroll
  for (int q = 6; q < 16; ++q) ow[q] = 0;
  uint4* o = (uint4*)(x0p + (size_t)gi * 32);
#pragma unroll
  for (int q = 0; q < 4; ++q)
    o[q] = make_uint4(ow[4 * q], ow[4 * q + 1], ow[4 * q + 2], ow[4 * q + 3]);
}

// ====== Kernel 3: MLP1 12->64->64->64 (relu), fused via per-wave LDS ========
__global__ void mlp1_kernel(const __bf16* __restrict__ x0p,
                            const __bf16* pw1, const float* b1,
                            const __bf16* pw2, const float* b2,
                            const __bf16* pw3, const float* b3,
                            __bf16* __restrict__ h3) {
  __shared__ alignas(16) __bf16 tile[8][16 * 72];
  const int TS = 72;
  int tid = threadIdx.x, lane = tid & 31, wave = tid >> 5;
  int row0 = blockIdx.x * 128 + wave * 16;
  __bf16* t = &tile[wave][0];

  bf16x16 ax = load_a_bf16(x0p, row0, 32, 0, lane);     // layer1 (K padded 32)
#pragma unroll
  for (int g = 0; g < 4; ++g) {
    f32x8 c = {};
    c = wmma_bf16(ax, load_b_packed(pw1, g, 1, 0, lane), c);
    stage_acc(t, TS, c, g * 16, lane, b1, true);
  }
  bf16x16 a0 = load_a_lds(t, TS, 0, lane);              // layer2: 64->64
  bf16x16 a1 = load_a_lds(t, TS, 32, lane);
  f32x8 acc2[4];
#pragma unroll
  for (int g = 0; g < 4; ++g) {
    f32x8 c = {};
    c = wmma_bf16(a0, load_b_packed(pw2, g, 2, 0, lane), c);
    c = wmma_bf16(a1, load_b_packed(pw2, g, 2, 1, lane), c);
    acc2[g] = c;
  }
#pragma unroll
  for (int g = 0; g < 4; ++g) stage_acc(t, TS, acc2[g], g * 16, lane, b2, true);
  a0 = load_a_lds(t, TS, 0, lane);                      // layer3: 64->64
  a1 = load_a_lds(t, TS, 32, lane);
#pragma unroll
  for (int g = 0; g < 4; ++g) {
    f32x8 c = {};
    c = wmma_bf16(a0, load_b_packed(pw3, g, 2, 0, lane), c);
    c = wmma_bf16(a1, load_b_packed(pw3, g, 2, 1, lane), c);
    acc2[g] = c;
  }
#pragma unroll
  for (int g = 0; g < 4; ++g) stage_acc(t, TS, acc2[g], g * 16, lane, b3, true);
  store_tile_rows<64>(t, TS, h3 + (size_t)row0 * 64, 64, lane);
}

// ====== Kernels 4/6: local max-pool over K=16 neighbors (bf16, uint4) =======
template <int C>
__global__ void maxpool_kernel(const __bf16* __restrict__ x, const int* __restrict__ idx,
                               __bf16* __restrict__ out) {
  const int CG = C / 8;
  int tg = blockIdx.x * 256 + threadIdx.x;
  if (tg >= Mtot * CG) return;
  int cg = tg % CG, gi = tg / CG, b = gi >> 12;
  const int4* nb4 = (const int4*)(idx + (size_t)gi * 16);
  int nb[16];
#pragma unroll
  for (int q = 0; q < 4; ++q) {
    int4 n4 = nb4[q];
    nb[4 * q] = n4.x; nb[4 * q + 1] = n4.y; nb[4 * q + 2] = n4.z; nb[4 * q + 3] = n4.w;
  }
  float mxv[8];
#pragma unroll
  for (int c = 0; c < 8; ++c) mxv[c] = -3.4e38f;
#pragma unroll 4
  for (int k = 0; k < 16; ++k) {
    size_t j = (size_t)b * Npts + nb[k];
    uint4 u = *(const uint4*)(x + j * C + cg * 8);
    unsigned w[4] = {u.x, u.y, u.z, u.w};
#pragma unroll
    for (int q = 0; q < 4; ++q) {
      mxv[2 * q]     = fmaxf(mxv[2 * q],     bflo(w[q]));
      mxv[2 * q + 1] = fmaxf(mxv[2 * q + 1], bfhi(w[q]));
    }
  }
  unsigned ow[4];
#pragma unroll
  for (int q = 0; q < 4; ++q)
    ow[q] = f2bf(mxv[2 * q]) | (f2bf(mxv[2 * q + 1]) << 16);
  *(uint4*)(out + (size_t)gi * C + cg * 8) = make_uint4(ow[0], ow[1], ow[2], ow[3]);
}

// ====== Kernel 5: graph layer 1: lin 64->64 then relu 64->128 ===============
__global__ void graph1_kernel(const __bf16* __restrict__ m1,
                              const __bf16* plw1, const float* lb1,
                              const __bf16* pcw1, const float* cb1,
                              __bf16* __restrict__ u) {
  __shared__ alignas(16) __bf16 tile[8][16 * 136];
  const int TS = 136;
  int tid = threadIdx.x, lane = tid & 31, wave = tid >> 5;
  int row0 = blockIdx.x * 128 + wave * 16;
  __bf16* t = &tile[wave][0];
  bf16x16 a0 = load_a_bf16(m1, row0, 64, 0, lane);
  bf16x16 a1 = load_a_bf16(m1, row0, 64, 32, lane);
#pragma unroll
  for (int g = 0; g < 4; ++g) {
    f32x8 c = {};
    c = wmma_bf16(a0, load_b_packed(plw1, g, 2, 0, lane), c);
    c = wmma_bf16(a1, load_b_packed(plw1, g, 2, 1, lane), c);
    stage_acc(t, TS, c, g * 16, lane, lb1, false);
  }
  a0 = load_a_lds(t, TS, 0, lane);
  a1 = load_a_lds(t, TS, 32, lane);
  f32x8 acc2[8];
#pragma unroll
  for (int g = 0; g < 8; ++g) {
    f32x8 c = {};
    c = wmma_bf16(a0, load_b_packed(pcw1, g, 2, 0, lane), c);
    c = wmma_bf16(a1, load_b_packed(pcw1, g, 2, 1, lane), c);
    acc2[g] = c;
  }
#pragma unroll
  for (int g = 0; g < 8; ++g) stage_acc(t, TS, acc2[g], g * 16, lane, cb1, true);
  store_tile_rows<128>(t, TS, u + (size_t)row0 * 128, 128, lane);
}

// ====== Kernel 7: graph layer 2 (128->128 -> 128->1024) + fused block max ===
// y-phase weights (256KB packed cw2) are async-DMA'd into double-buffered LDS
// once per block (shared by all 8 waves) and consumed via ds_load_b128.
__global__ void graph2_kernel(const __bf16* __restrict__ m2,
                              const __bf16* plw2, const float* lb2,
                              const __bf16* pcw2, float* __restrict__ partial) {
  __shared__ alignas(16) __bf16 tile[8][16 * 136];
  __shared__ alignas(16) __bf16 wbuf[2][2048];   // 2 x 4KB (4 frags x 32 lanes x 32B)
  __shared__ int ymax[1024];
  const int TS = 136;
  int tid = threadIdx.x, lane = tid & 31, wave = tid >> 5;
  int row0 = blockIdx.x * 128 + wave * 16;
  for (int c = tid; c < 1024; c += 256) ymax[c] = 0xFF800000;  // -inf
  __bf16* t = &tile[wave][0];
  bf16x16 a0 = load_a_bf16(m2, row0, 128, 0, lane);
  bf16x16 a1 = load_a_bf16(m2, row0, 128, 32, lane);
  bf16x16 a2 = load_a_bf16(m2, row0, 128, 64, lane);
  bf16x16 a3 = load_a_bf16(m2, row0, 128, 96, lane);
#pragma unroll
  for (int g = 0; g < 8; ++g) {   // v = m2 @ lw2^T + lb2 (no relu)
    f32x8 c = {};
    c = wmma_bf16(a0, load_b_packed(plw2, g, 4, 0, lane), c);
    c = wmma_bf16(a1, load_b_packed(plw2, g, 4, 1, lane), c);
    c = wmma_bf16(a2, load_b_packed(plw2, g, 4, 2, lane), c);
    c = wmma_bf16(a3, load_b_packed(plw2, g, 4, 3, lane), c);
    stage_acc(t, TS, c, g * 16, lane, lb2, false);
  }
  a0 = load_a_lds(t, TS, 0, lane);
  a1 = load_a_lds(t, TS, 32, lane);
  a2 = load_a_lds(t, TS, 64, lane);
  a3 = load_a_lds(t, TS, 96, lane);

  // async DMA of one 4KB fragment block (wave 0 only: 8 x 32 lanes x 16B)
  auto issue_copy = [&](int g, int buf) {
    unsigned dbase = lds_addr32(&wbuf[buf][0]);
    const char* sbase = (const char*)pcw2 + (size_t)g * 4096;
#pragma unroll
    for (int it = 0; it < 8; ++it) {
      int chunk = it * 32 + lane;
      async_copy_b128(dbase + chunk * 16, sbase + (size_t)chunk * 16);
    }
  };
  if (wave == 0) { issue_copy(0, 0); wait_async0(); }
  __syncthreads();                                 // buf0 ready, ymax init'd

  for (int g = 0; g < 64; ++g) {  // y = v @ cw2^T, fused max over 16-row tile
    int buf = g & 1;
    if (wave == 0 && g < 63) issue_copy(g + 1, buf ^ 1);   // overlap DMA
    const __bf16* wb = &wbuf[buf][0];
    f32x8 c = {};
    c = wmma_bf16(a0, load_b_lds(wb, 0, lane), c);
    c = wmma_bf16(a1, load_b_lds(wb, 1, lane), c);
    c = wmma_bf16(a2, load_b_lds(wb, 2, lane), c);
    c = wmma_bf16(a3, load_b_lds(wb, 3, lane), c);
    float mx = c[0];
#pragma unroll
    for (int r = 1; r < 8; ++r) mx = fmaxf(mx, c[r]);
    mx = fmaxf(mx, __shfl_xor(mx, 16, 32));        // combine row halves
    if (lane < 16) atomicMaxFloatShared(&ymax[g * 16 + lane], mx);
    if (wave == 0 && g < 63) wait_async0();
    __syncthreads();                               // publish next buffer
  }
  int b = blockIdx.x >> 5, slot = blockIdx.x & 31;
  for (int c = tid; c < 1024; c += 256)
    partial[((size_t)b * 32 + slot) * 1024 + c] = __int_as_float(ymax[c]);
}

// ====== Kernel 8: reduce partial maxes + MLP2 (1024->512 relu -> 512) =======
__global__ void tail_kernel(const float* __restrict__ partial, const float* cb2,
                            const float* mw1, const float* mb1,
                            const float* mw2, const float* mb2,
                            float* __restrict__ out) {
  __shared__ float g[1024];
  __shared__ float h[512];
  int b = blockIdx.x, tid = threadIdx.x;  // 512 threads
  for (int c = tid; c < 1024; c += 512) {
    float mx = -3.4e38f;
    for (int s = 0; s < 32; ++s)
      mx = fmaxf(mx, partial[((size_t)b * 32 + s) * 1024 + c]);
    g[c] = mx + cb2[c];
  }
  __syncthreads();
  float acc = mb1[tid];
  for (int k = 0; k < 1024; ++k) acc = fmaf(g[k], mw1[(size_t)tid * 1024 + k], acc);
  h[tid] = fmaxf(acc, 0.0f);
  __syncthreads();
  float acc2 = mb2[tid];
  for (int k = 0; k < 512; ++k) acc2 = fmaf(h[k], mw2[(size_t)tid * 512 + k], acc2);
  out[(size_t)b * 512 + tid] = acc2;
}

// ============================================================================
extern "C" void kernel_launch(void* const* d_in, const int* in_sizes, int n_in,
                              void* d_out, int out_size, void* d_ws, size_t ws_size,
                              hipStream_t stream) {
  const float* pts = (const float*)d_in[0];
  const float* w1  = (const float*)d_in[1];  const float* b1  = (const float*)d_in[2];
  const float* w2  = (const float*)d_in[3];  const float* b2  = (const float*)d_in[4];
  const float* w3  = (const float*)d_in[5];  const float* b3  = (const float*)d_in[6];
  const float* lw1 = (const float*)d_in[7];  const float* lb1 = (const float*)d_in[8];
  const float* cw1 = (const float*)d_in[9];  const float* cb1 = (const float*)d_in[10];
  const float* lw2 = (const float*)d_in[11]; const float* lb2 = (const float*)d_in[12];
  const float* cw2 = (const float*)d_in[13]; const float* cb2 = (const float*)d_in[14];
  const float* mw1 = (const float*)d_in[15]; const float* mb1 = (const float*)d_in[16];
  const float* mw2 = (const float*)d_in[17]; const float* mb2 = (const float*)d_in[18];

  // workspace layout:
  //   [0,2MB)    idx int [M,16]
  //   [2,10MB)   A: x0p bf16 [M,32] -> m1 bf16 [M,64] -> m2 bf16 [M,128]
  //   [10,18MB)  B: h3 bf16 [M,64] -> u bf16 [M,128]
  //   [18,19MB)  partial f32 [B,32,1024]
  //   [19MB,..)  packed weights (bf16 fragment layout, ~340KB)
  char* ws = (char*)d_ws;
  int*    idx     = (int*)ws;
  __bf16* x0p     = (__bf16*)(ws + (2u << 20));
  __bf16* m1      = (__bf16*)(ws + (2u << 20));
  __bf16* m2      = (__bf16*)(ws + (2u << 20));
  __bf16* h3      = (__bf16*)(ws + (10u << 20));
  __bf16* u       = (__bf16*)(ws + (10u << 20));
  float*  partial = (float*)(ws + (18u << 20));
  __bf16* pwbase  = (__bf16*)(ws + (19u << 20));
  __bf16* pw1  = pwbase;          // 4*1*512  = 2048 elems
  __bf16* pw2  = pw1  + 2048;     // 4*2*512  = 4096
  __bf16* pw3  = pw2  + 4096;     // 4096
  __bf16* plw1 = pw3  + 4096;     // 4096
  __bf16* pcw1 = plw1 + 4096;     // 8*2*512  = 8192
  __bf16* plw2 = pcw1 + 8192;     // 8*4*512  = 16384
  __bf16* pcw2 = plw2 + 16384;    // 64*4*512 = 131072
  float*  out  = (float*)d_out;

  auto pack = [&](const float* W, __bf16* dst, int Cin, int nG, int nKC) {
    int tot = nG * nKC * 32;
    pack_w_kernel<<<(tot + 255) / 256, 256, 0, stream>>>(W, dst, Cin, nG, nKC);
  };
  pack(w1,  pw1,  12,  4, 1);
  pack(w2,  pw2,  64,  4, 2);
  pack(w3,  pw3,  64,  4, 2);
  pack(lw1, plw1, 64,  4, 2);
  pack(cw1, pcw1, 64,  8, 2);
  pack(lw2, plw2, 128, 8, 4);
  pack(cw2, pcw2, 128, 64, 4);

  knn_kernel<<<dim3(Npts / 256, Bsz), 256, 0, stream>>>(pts, idx);
  cov_kernel<<<Mtot / 256, 256, 0, stream>>>(pts, idx, x0p);
  mlp1_kernel<<<Mtot / 128, 256, 0, stream>>>(x0p, pw1, b1, pw2, b2, pw3, b3, h3);
  maxpool_kernel<64><<<(Mtot * 8 + 255) / 256, 256, 0, stream>>>(h3, idx, m1);
  graph1_kernel<<<Mtot / 128, 256, 0, stream>>>(m1, plw1, lb1, pcw1, cb1, u);
  maxpool_kernel<128><<<(Mtot * 16 + 255) / 256, 256, 0, stream>>>(u, idx, m2);
  graph2_kernel<<<Mtot / 128, 256, 0, stream>>>(m2, plw2, lb2, pcw2, partial);
  tail_kernel<<<Bsz, 512, 0, stream>>>(partial, cb2, mw1, mb1, mw2, mb2, out);
}
